// Multihead_Latent_Attention_36464272343184
// MI455X (gfx1250) — matrix operand order repaired and checked by reading, not compiled, not run on hardware
//
#include <hip/hip_runtime.h>
#include <hip/hip_bf16.h>

// ---- MLA problem constants (match reference) ----
constexpr int H   = 2048;
constexpr int NH  = 16;
constexpr int QL  = 1536;
constexpr int KVL = 512;
constexpr int QR  = 64;
constexpr int QN  = 128;
constexpr int KR  = 64;
constexpr int KN  = 128;
constexpr int VD  = 128;
constexpr int S   = 2048;
constexpr int KVD = KVL + KR;          // 576
constexpr int QUP = NH * (QN + QR);    // 3072
constexpr int KUP = NH * (KN + VD);    // 4096
constexpr int QKD = KR + QN;           // 192 (rope||nope)

typedef __attribute__((ext_vector_type(16))) __bf16 v16bf;
typedef __attribute__((ext_vector_type(8)))  __bf16 v8bf;
typedef __attribute__((ext_vector_type(8)))  float  v8f;

// ---- fragment loaders -------------------------------------------------------
// A-matrix fragment (16x32 bf16): lane-half selects K {0..7,16..23} vs {8..15,24..31}
__device__ __forceinline__ v16bf frag_a(const __bf16* __restrict__ p, int half) {
  const v8bf lo = *(const v8bf*)(p + (half ? 8 : 0));
  const v8bf hi = *(const v8bf*)(p + (half ? 24 : 16));
  v16bf r;
#pragma unroll
  for (int i = 0; i < 8; ++i) { r[i] = lo[i]; r[8 + i] = hi[i]; }
  return r;
}
__device__ __forceinline__ v16bf frag_a(const float* __restrict__ p, int half) {
  const v8f lo = *(const v8f*)(p + (half ? 8 : 0));
  const v8f hi = *(const v8f*)(p + (half ? 24 : 16));
  v16bf r;
#pragma unroll
  for (int i = 0; i < 8; ++i) { r[i] = (__bf16)lo[i]; r[8 + i] = (__bf16)hi[i]; }
  return r;
}
// B-matrix fragment (32x16, operand = W^T from row-major W):
// lane-half selects K {0..7,16..23} vs {8..15,24..31}
__device__ __forceinline__ v16bf frag_b(const __bf16* __restrict__ p, int half) {
  const v8bf lo = *(const v8bf*)(p + (half ? 8 : 0));
  const v8bf hi = *(const v8bf*)(p + (half ? 24 : 16));
  v16bf r;
#pragma unroll
  for (int i = 0; i < 8; ++i) { r[i] = lo[i]; r[8 + i] = hi[i]; }
  return r;
}
__device__ __forceinline__ v16bf frag_b(const float* __restrict__ p, int half) {
  const v8f lo = *(const v8f*)(p + (half ? 8 : 0));
  const v8f hi = *(const v8f*)(p + (half ? 24 : 16));
  v16bf r;
#pragma unroll
  for (int i = 0; i < 8; ++i) { r[i] = (__bf16)lo[i]; r[8 + i] = (__bf16)hi[i]; }
  return r;
}

__device__ __forceinline__ v8f wmma_bf16(v16bf a, v16bf b, v8f c) {
  // (neg_a, A, neg_b, B, c_mod, C, reuse_a, reuse_b)
  return __builtin_amdgcn_wmma_f32_16x16x32_bf16(false, a, false, b, (short)0, c,
                                                 false, false);
}

// ---- batched GEMM: C[M,N] = A[M,K] * B[N,K]^T ------------------------------
// Block = 128 threads (4 waves stacked in M); block tile 128x64.
// Wave tile 32x64: 2 A-frags x 4 B-frags -> 8 WMMAs per K-step.
// Grid: x = N/64, y = M/128, z = batch.
template <typename TA, typename TB, typename TC>
__global__ void __launch_bounds__(128)
gemm_bt(const TA* __restrict__ A, const TB* __restrict__ B, TC* __restrict__ C,
        int K, int lda, int ldb, int ldc,
        long long sA, long long sB, long long sC) {
  A += (long long)blockIdx.z * sA;
  B += (long long)blockIdx.z * sB;
  C += (long long)blockIdx.z * sC;

  const int lane = threadIdx.x & 31;
  const int wave = threadIdx.x >> 5;
  const int half = lane >> 4;
  const int idx  = lane & 15;
  const int tm   = blockIdx.y * 128 + wave * 32;
  const int tn   = blockIdx.x * 64;

  const TA* a0 = A + (long long)(tm + idx) * lda;
  const TA* a1 = a0 + 16LL * lda;
  const TB* b0 = B + (long long)(tn + idx) * ldb;
  const TB* b1 = b0 + 16LL * ldb;
  const TB* b2 = b1 + 16LL * ldb;
  const TB* b3 = b2 + 16LL * ldb;

  v8f acc[2][4];
#pragma unroll
  for (int i = 0; i < 2; ++i)
#pragma unroll
    for (int j = 0; j < 4; ++j) acc[i][j] = (v8f){};

  for (int k = 0; k < K; k += 32) {
    if (k + 64 < K) {  // global_prefetch next K tile (hits L2)
      __builtin_prefetch(a0 + k + 64, 0, 1);
      __builtin_prefetch(b0 + k + 64, 0, 1);
    }
    const v16bf fa0 = frag_a(a0 + k, half);
    const v16bf fa1 = frag_a(a1 + k, half);
    const v16bf fb0 = frag_b(b0 + k, half);
    const v16bf fb1 = frag_b(b1 + k, half);
    const v16bf fb2 = frag_b(b2 + k, half);
    const v16bf fb3 = frag_b(b3 + k, half);
    acc[0][0] = wmma_bf16(fa0, fb0, acc[0][0]);
    acc[0][1] = wmma_bf16(fa0, fb1, acc[0][1]);
    acc[0][2] = wmma_bf16(fa0, fb2, acc[0][2]);
    acc[0][3] = wmma_bf16(fa0, fb3, acc[0][3]);
    acc[1][0] = wmma_bf16(fa1, fb0, acc[1][0]);
    acc[1][1] = wmma_bf16(fa1, fb1, acc[1][1]);
    acc[1][2] = wmma_bf16(fa1, fb2, acc[1][2]);
    acc[1][3] = wmma_bf16(fa1, fb3, acc[1][3]);
  }

#pragma unroll
  for (int i = 0; i < 2; ++i) {
#pragma unroll
    for (int r = 0; r < 8; ++r) {
      const long long row = tm + i * 16 + half * 8 + r;
#pragma unroll
      for (int j = 0; j < 4; ++j)
        C[row * ldc + tn + j * 16 + idx] = (TC)acc[i][j][r];
    }
  }
}

// ---- elementwise f32 -> bf16 (8 elements / thread, vectorized) -------------
__global__ void __launch_bounds__(256)
cvt_bf16_kernel(const float* __restrict__ in, __bf16* __restrict__ out, long long n) {
  long long i = ((long long)blockIdx.x * blockDim.x + threadIdx.x) * 8;
  if (i + 8 <= n) {
    const v8f v = *(const v8f*)(in + i);
    v8bf o;
#pragma unroll
    for (int j = 0; j < 8; ++j) o[j] = (__bf16)v[j];
    *(v8bf*)(out + i) = o;
  } else {
    for (; i < n; ++i) out[i] = (__bf16)in[i];
  }
}

// ---- RMSNorm over rows of length D; f32 in, bf16 out (one block per row) ---
__global__ void __launch_bounds__(256)
rmsnorm_kernel(const float* __restrict__ x, const float* __restrict__ w,
               __bf16* __restrict__ y, int D) {
  const float* p = x + (long long)blockIdx.x * D;
  __bf16* q = y + (long long)blockIdx.x * D;
  __shared__ float red[256];
  const int tid = threadIdx.x;
  float s = 0.f;
  for (int c = tid; c < D; c += 256) { const float v = p[c]; s += v * v; }
  red[tid] = s; __syncthreads();
  for (int off = 128; off > 0; off >>= 1) {
    if (tid < off) red[tid] += red[tid + off];
    __syncthreads();
  }
  const float scale = rsqrtf(red[0] / (float)D + 1e-6f);
  for (int c = tid; c < D; c += 256) q[c] = (__bf16)(w[c] * p[c] * scale);
}

// ---- RoPE + head-major assembly: writes bf16 query/key (rope||nope, d=192)
// and bf16 value^T [NH][VD][S] so PV is also an A*B^T GEMM. -----------------
__global__ void __launch_bounds__(192)
assemble_kernel(const float* __restrict__ qbuf, const float* __restrict__ kvbuf,
                const __bf16* __restrict__ ckv, __bf16* __restrict__ query,
                __bf16* __restrict__ key, __bf16* __restrict__ valT) {
  const int s = blockIdx.x, h = blockIdx.y, j = threadIdx.x;  // j in [0,192)
  const float* qrow  = qbuf  + (long long)s * QUP + h * (QN + QR);
  const float* kvrow = kvbuf + (long long)s * KUP + h * (KN + VD);
  const __bf16* kpe  = ckv   + (long long)s * KVD + KVL;

  float qv, kv;
  if (j < KR) {
    const int i = j & 31;
    const float inv_freq = __expf(-(2.0f * (float)i / 64.0f) * 9.210340371976184f); // ln(1e4)
    const float ang = (float)s * inv_freq;
    const float c = __cosf(ang), sn = __sinf(ang);
    const float q0 = qrow[QN + 2 * i], q1 = qrow[QN + 2 * i + 1];
    const float k0 = (float)kpe[2 * i], k1 = (float)kpe[2 * i + 1];
    if (j < 32) { qv = q0 * c - q1 * sn;  kv = k0 * c - k1 * sn; }
    else        { qv = q1 * c + q0 * sn;  kv = k1 * c + k0 * sn; }
  } else {
    qv = qrow[j - KR];
    kv = kvrow[j - KR];
  }
  const long long o = ((long long)h * S + s) * QKD + j;
  query[o] = (__bf16)qv;
  key[o]   = (__bf16)kv;
  if (j < VD) valT[((long long)h * VD + j) * S + s] = (__bf16)kvrow[KN + j];
}

// ---- in-place causal softmax (one logits row per block); masked tail -> 0 --
__global__ void __launch_bounds__(256)
softmax_causal(float* __restrict__ logits) {
  const int q = blockIdx.x, h = blockIdx.y;
  float* row = logits + ((long long)h * S + q) * S;
  __shared__ float red[256];
  const int tid = threadIdx.x;
  const int valid = q + 1;

  float m = -INFINITY;
  for (int k = tid; k < valid; k += 256) m = fmaxf(m, row[k]);
  red[tid] = m; __syncthreads();
  for (int off = 128; off > 0; off >>= 1) {
    if (tid < off) red[tid] = fmaxf(red[tid], red[tid + off]);
    __syncthreads();
  }
  m = red[0]; __syncthreads();

  float sum = 0.f;
  for (int k = tid; k < valid; k += 256) {
    const float e = __expf(row[k] - m);
    row[k] = e; sum += e;
  }
  red[tid] = sum; __syncthreads();
  for (int off = 128; off > 0; off >>= 1) {
    if (tid < off) red[tid] += red[tid + off];
    __syncthreads();
  }
  const float inv = 1.0f / red[0];
  for (int k = tid; k < valid; k += 256) row[k] *= inv;
  for (int k = valid + tid; k < S; k += 256) row[k] = 0.f;
}

extern "C" void kernel_launch(void* const* d_in, const int* in_sizes, int n_in,
                              void* d_out, int out_size, void* d_ws, size_t ws_size,
                              hipStream_t stream) {
  (void)in_sizes; (void)n_in; (void)out_size; (void)ws_size;
  const float* hidden    = (const float*)d_in[0];
  // d_in[1] position_ids / d_in[2] attn_mask: arange + causal tril by construction
  const float* q_down_w  = (const float*)d_in[3];
  const float* kv_down_w = (const float*)d_in[4];
  const float* q_up_w    = (const float*)d_in[5];
  const float* k_up_w    = (const float*)d_in[6];
  const float* o_w       = (const float*)d_in[7];
  const float* q_norm_w  = (const float*)d_in[8];
  const float* kv_norm_w = (const float*)d_in[9];
  float* out = (float*)d_out;

  char* ws = (char*)d_ws;
  auto allocF = [&](size_t n) -> float* {
    float* p = (float*)ws;
    ws += ((n * sizeof(float) + 255) / 256) * 256;
    return p;
  };
  auto allocB = [&](size_t n) -> __bf16* {
    __bf16* p = (__bf16*)ws;
    ws += ((n * sizeof(__bf16) + 255) / 256) * 256;
    return p;
  };

  // bf16 copies of all GEMM operands (all fit in the 192MB L2 together)
  __bf16* hid16 = allocB((size_t)S * H);
  __bf16* qdw16 = allocB((size_t)QL * H);
  __bf16* kvdw16 = allocB((size_t)KVD * H);
  __bf16* qup16 = allocB((size_t)QUP * QL);
  __bf16* kup16 = allocB((size_t)KUP * KVL);
  __bf16* ow16  = allocB((size_t)H * (NH * VD));
  // intermediates
  float*  cqf   = allocF((size_t)S * QL);       // latent q (pre-norm, f32)
  float*  ckvf  = allocF((size_t)S * KVD);      // compressed kv (pre-norm, f32)
  __bf16* cq16  = allocB((size_t)S * QL);       // normed latent q
  __bf16* ckv16 = allocB((size_t)S * KVD);      // normed compressed kv
  float*  qbuf  = allocF((size_t)S * QUP);      // q up-proj
  float*  kvbuf = allocF((size_t)S * KUP);      // kv up-proj
  __bf16* qry16 = allocB((size_t)NH * S * QKD); // head-major rope||nope
  __bf16* key16 = allocB((size_t)NH * S * QKD);
  __bf16* valT16 = allocB((size_t)NH * VD * S); // value transposed per head
  __bf16* acat16 = allocB((size_t)S * NH * VD); // concat attn output

  const dim3 blk(128);
  auto cvt = [&](const float* src, __bf16* dst, long long n) {
    cvt_bf16_kernel<<<(unsigned)((n / 8 + 255) / 256), 256, 0, stream>>>(src, dst, n);
  };

  // 0) one-shot bf16 conversion of all operands
  cvt(hidden,    hid16,  (long long)S * H);
  cvt(q_down_w,  qdw16,  (long long)QL * H);
  cvt(kv_down_w, kvdw16, (long long)KVD * H);
  cvt(q_up_w,    qup16,  (long long)QUP * QL);
  cvt(k_up_w,    kup16,  (long long)KUP * KVL);
  cvt(o_w,       ow16,   (long long)H * (NH * VD));

  // 1) down-projections (K = H)
  gemm_bt<__bf16, __bf16, float><<<dim3(QL / 64, S / 128, 1), blk, 0, stream>>>(
      hid16, qdw16, cqf, H, H, H, QL, 0, 0, 0);
  gemm_bt<__bf16, __bf16, float><<<dim3(KVD / 64, S / 128, 1), blk, 0, stream>>>(
      hid16, kvdw16, ckvf, H, H, H, KVD, 0, 0, 0);

  // 2) RMSNorm -> bf16
  rmsnorm_kernel<<<S, 256, 0, stream>>>(cqf, q_norm_w, cq16, QL);
  rmsnorm_kernel<<<S, 256, 0, stream>>>(ckvf, kv_norm_w, ckv16, KVD);

  // 3) up-projections
  gemm_bt<__bf16, __bf16, float><<<dim3(QUP / 64, S / 128, 1), blk, 0, stream>>>(
      cq16, qup16, qbuf, QL, QL, QL, QUP, 0, 0, 0);
  gemm_bt<__bf16, __bf16, float><<<dim3(KUP / 64, S / 128, 1), blk, 0, stream>>>(
      ckv16, kup16, kvbuf, KVL, KVD, KVL, KUP, 0, 0, 0);

  // 4) RoPE + head-major assembly (writes bf16 operands for attention GEMMs)
  assemble_kernel<<<dim3(S, NH), 192, 0, stream>>>(qbuf, kvbuf, ckv16,
                                                   qry16, key16, valT16);

  // 5) logits = query @ key^T per head, straight into d_out (attn_weights slot)
  gemm_bt<__bf16, __bf16, float><<<dim3(S / 64, S / 128, NH), blk, 0, stream>>>(
      qry16, key16, out, QKD, QKD, QKD, S,
      (long long)S * QKD, (long long)S * QKD, (long long)S * S);

  // 6) causal softmax in place on d_out
  softmax_causal<<<dim3(S, NH), 256, 0, stream>>>(out);

  // 7) attn = P @ V per head (P read as f32 from d_out, cvt in-register)
  gemm_bt<float, __bf16, __bf16><<<dim3(VD / 64, S / 128, NH), blk, 0, stream>>>(
      out, valT16, acat16, S, S, S, NH * VD,
      (long long)S * S, (long long)VD * S, (long long)VD);

  // 8) output projection into d_out tail
  gemm_bt<__bf16, __bf16, float><<<dim3(H / 64, S / 128, 1), blk, 0, stream>>>(
      acat16, ow16, out + (long long)NH * S * S, NH * VD, NH * VD, NH * VD, H,
      0, 0, 0);
}
